// SeqLSTM_60765197304274
// MI455X (gfx1250) — matrix operand order repaired and checked by reading, not compiled, not run on hardware
//
#include <hip/hip_runtime.h>
#include <hip/hip_bf16.h>

typedef __attribute__((ext_vector_type(16))) _Float16 v16h;
typedef __attribute__((ext_vector_type(8)))  _Float16 v8h;
typedef __attribute__((ext_vector_type(4)))  _Float16 v4h;
typedef __attribute__((ext_vector_type(8)))  float    v8f;
typedef __attribute__((ext_vector_type(4)))  unsigned int u32x4;
typedef __attribute__((ext_vector_type(8)))  int      i32x8;
typedef __attribute__((ext_vector_type(4)))  int      i32x4;

#define B_   128
#define T_   256
#define E_   64
#define H_   128
#define D_   128     // 2E
#define G4_  512     // 4H
#define OUTN 256     // OUT*T
#define KOUT (T_*H_) // 32768
#define EPSL 1e-5f

__device__ __forceinline__ float sigmoidf_(float x) { return 1.0f / (1.0f + __expf(-x)); }

__device__ __forceinline__ v8f wmma_f16(v16h a, v16h b, v8f c) {
    // D = A*B + C, f16 inputs, f32 accumulate (v_wmma_f32_16x16x32_f16)
    return __builtin_amdgcn_wmma_f32_16x16x32_f16(false, a, false, b, (short)0, c, false, false);
}

// A fragment (16x32, M x K). lanes 0-15 -> K 0..15, lanes 16-31 -> K 16..31.
__device__ __forceinline__ v16h load_a_frag(const _Float16* A, int lda, int k0) {
    const int lane = threadIdx.x & 31;
    const int row  = lane & 15;
    const int kb   = k0 + ((lane & 16) ? 16 : 0);
    const v8h* p = (const v8h*)(A + row * lda + kb);
    v8h lo = p[0];   // K = kb .. kb+7
    v8h hi = p[1];   // K = kb+8 .. kb+15
    v16h r;
#pragma unroll
    for (int i = 0; i < 8; ++i) { r[i] = lo[i]; r[8 + i] = hi[i]; }
    return r;
}

// B fragment (32x16, K x N) with B = W^T, W row-major (N x K, leading dim ldw).
__device__ __forceinline__ v16h load_b_frag(const _Float16* W, int ldw, int k0) {
    const int lane = threadIdx.x & 31;
    const int row  = lane & 15;
    const int kb   = k0 + ((lane & 16) ? 16 : 0);
    const v8h* p = (const v8h*)(W + row * ldw + kb);
    v8h lo = p[0];
    v8h hi = p[1];
    v16h r;
#pragma unroll
    for (int i = 0; i < 8; ++i) { r[i] = lo[i]; r[8 + i] = hi[i]; }
    return r;
}

// ---- Tensor Data Mover: 2D tile load (rows=16, rowlen=512 f32, row stride in elems)
// D# per CDNA5 ISA. Issued by one wave; tracked by TENSORcnt.
// This toolchain exposes the 6-arg builtin (extra int32x8 SGPR group, unused for 2D).
__device__ __forceinline__ void tdm_load_tile_16x512(const float* gsrc, unsigned lds_off,
                                                     unsigned row_stride_elems) {
    const unsigned long long ga = (unsigned long long)gsrc;
    u32x4 g0;
    g0[0] = 1u;                                   // count=1, user desc
    g0[1] = lds_off;                              // lds_addr (bytes)
    g0[2] = (unsigned)(ga & 0xffffffffu);         // global_addr[31:0]
    g0[3] = (unsigned)((ga >> 32) & 0x1ffffffu)   // global_addr[56:32]
            | (2u << 30);                         // type = 2 ("image")
    i32x8 g1;
    g1[0] = (2 << 16);                            // workgroup_mask=0, data_size=2 (4B)
    g1[1] = (int)(512u << 16);                    // tensor_dim0[15:0] at bit48 = 512
    g1[2] = (16 << 16);                           // tensor_dim0[31:16]=0 | tensor_dim1=16
    g1[3] = (int)(512u << 16);                    // tile_dim0 = 512
    g1[4] = 16;                                   // tile_dim1 = 16, tile_dim2 = 0
    g1[5] = (int)row_stride_elems;                // tensor_dim0_stride[31:0]
    g1[6] = 0;                                    // stride[47:32]=0, dim1_stride lo=0
    g1[7] = 0;
    const i32x4 z4 = {0, 0, 0, 0};                // groups 2/3 unused (<=2D tensor)
    const i32x8 z8 = {0, 0, 0, 0, 0, 0, 0, 0};    // extra group (6-arg form), unused
    __builtin_amdgcn_tensor_load_to_lds(g0, g1, z4, z4, z8, 0);
}

// ---------------- Kernel 1: embedding gather (also produces tuple output #2) ----
__global__ void gather_embed_kernel(const long long* __restrict__ idx,
                                    const float* __restrict__ table,
                                    float* __restrict__ emb_out,      // (B*T,128) f32 -> d_out
                                    _Float16* __restrict__ emb16) {   // f16 copy for WMMA
    const int tid = blockIdx.x * blockDim.x + threadIdx.x;  // B*T*2*16 threads (float4)
    const int e4 = tid & 15;
    const int j  = (tid >> 4) & 1;
    const int bt = tid >> 5;
    const long long g = idx[(long long)bt * 2 + j];
    const float4 v = ((const float4*)table)[g * 16 + e4];
    const int o = bt * 32 + j * 16 + e4;
    ((float4*)emb_out)[o] = v;
    v4h hv = { (_Float16)v.x, (_Float16)v.y, (_Float16)v.z, (_Float16)v.w };
    ((v4h*)emb16)[o] = hv;
}

// ---------------- f32 -> f16 weight conversion --------------------------------
__global__ void f32_to_f16_kernel(const float* __restrict__ src,
                                  _Float16* __restrict__ dst, int n) {
    const int i = blockIdx.x * blockDim.x + threadIdx.x;
    if (i < n) dst[i] = (_Float16)src[i];
}

// ---------------- Kernel 2: x_gates = emb @ W_ih^T  (32768x512, K=128) --------
__global__ void __launch_bounds__(256)
xgates_gemm_kernel(const _Float16* __restrict__ emb16,
                   const _Float16* __restrict__ Wih16,   // (512,128) row-major
                   float* __restrict__ xg) {             // (B*T, 512) f32
    const int wave = threadIdx.x >> 5;
    const int lane = threadIdx.x & 31;
    const int tile = blockIdx.x * 8 + wave;   // 65536 tiles of 16x16
    const int mt = tile >> 5;
    const int nt = tile & 31;
    const _Float16* At = emb16 + (size_t)mt * 16 * D_;
    const _Float16* Bt = Wih16 + (size_t)nt * 16 * D_;
    v8f acc = {};
#pragma unroll
    for (int ks = 0; ks < 4; ++ks) {
        v16h a = load_a_frag(At, D_, ks * 32);
        v16h b = load_b_frag(Bt, D_, ks * 32);
        acc = wmma_f16(a, b, acc);
    }
    const int n  = nt * 16 + (lane & 15);
    const int mb = mt * 16 + ((lane & 16) ? 8 : 0);
#pragma unroll
    for (int r = 0; r < 8; ++r)
        xg[(size_t)(mb + r) * G4_ + n] = acc[r];
}

// ---------------- Kernel 3: persistent LSTM scan ------------------------------
// 8 workgroups x 256 threads; each owns 16 batch rows for all 256 steps.
// W_hh fragments live in VGPRs for the whole recurrence (16 frags/wave = 128 VGPRs).
// x_gates tiles are DMA'd in by the Tensor Data Mover one step ahead (double buffer).
__global__ void __launch_bounds__(256, 1)
lstm_scan_kernel(const float* __restrict__ xg,
                 const float* __restrict__ Whh,     // (512,128) f32
                 const float* __restrict__ h0, const float* __restrict__ c0,
                 const float* __restrict__ g_ih, const float* __restrict__ b_ih,
                 const float* __restrict__ g_hh, const float* __restrict__ b_hh,
                 const float* __restrict__ g_c,  const float* __restrict__ b_c,
                 _Float16* __restrict__ hs16) {     // [T][B][H] f16
    __shared__ __align__(16) float    sxg[2 * 16 * G4_];  // 64 KB: TDM double buffer
    __shared__ __align__(16) float    sg[16 * G4_];       // 32 KB: gate scratch
    __shared__ __align__(16) _Float16 sh[16 * H_];        // 4 KB : h (f16)
    __shared__ __align__(16) float    sc[16 * H_];        // 8 KB : c (f32)

    const int tid  = threadIdx.x;
    const int b0   = blockIdx.x * 16;
    const int wave = tid >> 5;
    const int lane = tid & 31;

    // ---- hoist W_hh into per-wave register fragments (f32 -> f16, once) ----
    v16h bfr[4][4];
#pragma unroll
    for (int nti = 0; nti < 4; ++nti) {
        const int nt = wave * 4 + nti;
#pragma unroll
        for (int ks = 0; ks < 4; ++ks) {
            const int row = nt * 16 + (lane & 15);
            const int kb  = ks * 32 + ((lane & 16) ? 16 : 0);
            const float* p = Whh + row * H_ + kb;
            v16h b;
#pragma unroll
            for (int i = 0; i < 16; ++i) b[i] = (_Float16)p[i];
            bfr[nti][ks] = b;
        }
    }

    for (int i = tid; i < 16 * H_; i += 256) {
        sh[i] = (_Float16)h0[b0 * H_ + i];
        sc[i] = c0[b0 * H_ + i];
    }

    const unsigned sxg_off = (unsigned)(unsigned long long)(const void*)sxg;
    // prefetch x_gates tile for t=0 (wave 0 drives the TDM)
    if (wave == 0)
        tdm_load_tile_16x512(xg + (size_t)b0 * T_ * G4_, sxg_off, T_ * G4_);
    __syncthreads();

    const int lr = tid >> 4;           // row 0..15 (16 threads per row)
    const int lc = (tid & 15) * 32;    // 32 gate columns per thread
    const int j0 = (tid & 15) * 8;     // 8 cell elements per thread

    for (int t = 0; t < T_; ++t) {
        // ---- async prefetch of x_gates tile for t+1 (overlaps the GEMM) ----
        if (wave == 0 && t + 1 < T_)
            tdm_load_tile_16x512(xg + ((size_t)b0 * T_ + (t + 1)) * G4_,
                                 sxg_off + ((unsigned)((t + 1) & 1)) * 16 * G4_ * 4,
                                 T_ * G4_);

        // ---- hW = h @ W_hh^T : 16x512, K=128; W_hh from registers ----
        v16h afr[4];
#pragma unroll
        for (int ks = 0; ks < 4; ++ks) afr[ks] = load_a_frag(sh, H_, ks * 32);
#pragma unroll
        for (int nti = 0; nti < 4; ++nti) {
            v8f acc = {};
#pragma unroll
            for (int ks = 0; ks < 4; ++ks) acc = wmma_f16(afr[ks], bfr[nti][ks], acc);
            const int n  = (wave * 4 + nti) * 16 + (lane & 15);
            const int mb = (lane & 16) ? 8 : 0;
#pragma unroll
            for (int r = 0; r < 8; ++r) sg[(mb + r) * G4_ + n] = acc[r];
        }
        // retire the TDM load for step t (in-order: <=1 leaves only t+1 in flight)
        if (wave == 0) {
            if (t + 1 < T_) __builtin_amdgcn_s_wait_tensorcnt(1);
            else            __builtin_amdgcn_s_wait_tensorcnt(0);
        }
        __syncthreads();

        // ---- gates = LN(xg) + LN(hW), over the 512 dim per row ----
        const float* xrow = sxg + (t & 1) * 16 * G4_ + lr * G4_;
        float xv[32], hv[32];
        float sx = 0.f, sxx = 0.f, sh1 = 0.f, sh2 = 0.f;
#pragma unroll
        for (int i = 0; i < 32; ++i) {
            const float xi = xrow[lc + i];
            const float hi = sg[lr * G4_ + lc + i];
            xv[i] = xi; hv[i] = hi;
            sx += xi; sxx += xi * xi;
            sh1 += hi; sh2 += hi * hi;
        }
#pragma unroll
        for (int m = 1; m < 16; m <<= 1) {
            sx  += __shfl_xor(sx,  m, 32);
            sxx += __shfl_xor(sxx, m, 32);
            sh1 += __shfl_xor(sh1, m, 32);
            sh2 += __shfl_xor(sh2, m, 32);
        }
        const float mx = sx  * (1.f / G4_), mh = sh1 * (1.f / G4_);
        const float rx = rsqrtf(sxx * (1.f / G4_) - mx * mx + EPSL);
        const float rh = rsqrtf(sh2 * (1.f / G4_) - mh * mh + EPSL);
#pragma unroll
        for (int i = 0; i < 32; ++i) {
            const int col = lc + i;
            const float a  = (xv[i] - mx) * rx * g_ih[col] + b_ih[col];
            const float bb = (hv[i] - mh) * rh * g_hh[col] + b_hh[col];
            sg[lr * G4_ + col] = a + bb;
        }
        __syncthreads();

        // ---- cell/hidden update + LN(c_new) over H=128 ----
        float cn[8], ov[8];
        float ss = 0.f, ss2 = 0.f;
#pragma unroll
        for (int i = 0; i < 8; ++i) {
            const int j = j0 + i;
            const float ig = sg[lr * G4_ + j];
            const float fg = sg[lr * G4_ + H_ + j];
            const float gg = sg[lr * G4_ + 2 * H_ + j];
            const float og = sg[lr * G4_ + 3 * H_ + j];
            const float cc = sigmoidf_(fg) * sc[lr * H_ + j] + sigmoidf_(ig) * tanhf(gg);
            cn[i] = cc; ov[i] = og;
            ss += cc; ss2 += cc * cc;
        }
#pragma unroll
        for (int m = 1; m < 16; m <<= 1) {
            ss  += __shfl_xor(ss,  m, 32);
            ss2 += __shfl_xor(ss2, m, 32);
        }
        const float mc = ss * (1.f / H_);
        const float rc = rsqrtf(ss2 * (1.f / H_) - mc * mc + EPSL);
        _Float16* hsrow = hs16 + (size_t)t * B_ * H_ + (size_t)(b0 + lr) * H_;
#pragma unroll
        for (int i = 0; i < 8; ++i) {
            const int j = j0 + i;
            const float lnc = (cn[i] - mc) * rc * g_c[j] + b_c[j];
            const float hn  = sigmoidf_(ov[i]) * tanhf(lnc);
            sc[lr * H_ + j] = cn[i];
            sh[lr * H_ + j] = (_Float16)hn;
            hsrow[j] = (_Float16)hn;
        }
        __syncthreads();
    }
}

// ---------------- Kernel 4: out = sigmoid(seq @ W_out^T + b_out) --------------
// seq[b, t*H+h] = hs16[t][b][h]; 128x256 output, K=32768 streamed once.
__global__ void __launch_bounds__(256)
out_gemm_kernel(const _Float16* __restrict__ hs16,    // [T][B][H]
                const _Float16* __restrict__ Wout16,  // (256, 32768) row-major
                const float* __restrict__ b_out,
                float* __restrict__ out) {            // (128, 256)
    const int wave = threadIdx.x >> 5;
    const int lane = threadIdx.x & 31;
    const int tile = blockIdx.x * 8 + wave;  // 128 tiles of 16x16
    const int mt = tile >> 4;
    const int nt = tile & 15;
    const int m0 = mt * 16;
    const _Float16* Bt = Wout16 + (size_t)nt * 16 * KOUT;
    const int arow = lane & 15;
    const int kadj = (lane & 16) ? 16 : 0;
    v8f acc = {};
    for (int k0 = 0; k0 < KOUT; k0 += 32) {
        const _Float16* ap = hs16 + (size_t)(k0 >> 7) * B_ * H_
                                  + (size_t)(m0 + arow) * H_ + (k0 & (H_ - 1)) + kadj;
        const v8h* pa = (const v8h*)ap;
        v8h lo = pa[0], hi = pa[1];
        v16h a;
#pragma unroll
        for (int i = 0; i < 8; ++i) { a[i] = lo[i]; a[8 + i] = hi[i]; }
        v16h b = load_b_frag(Bt, KOUT, k0);
        __builtin_prefetch(Bt + (size_t)arow * KOUT + k0 + 1024, 0, 0);
        acc = wmma_f16(a, b, acc);
    }
    const int n  = nt * 16 + (lane & 15);
    const int mb = m0 + ((lane & 16) ? 8 : 0);
    const float bo = b_out[n];
#pragma unroll
    for (int r = 0; r < 8; ++r)
        out[(mb + r) * OUTN + n] = sigmoidf_(acc[r] + bo);
}

// ------------------------------------------------------------------------------
extern "C" void kernel_launch(void* const* d_in, const int* in_sizes, int n_in,
                              void* d_out, int out_size, void* d_ws, size_t ws_size,
                              hipStream_t stream) {
    const long long* indices = (const long long*)d_in[0];
    const float* h0    = (const float*)d_in[1];
    const float* c0    = (const float*)d_in[2];
    const float* table = (const float*)d_in[3];
    const float* W_ih  = (const float*)d_in[4];
    const float* W_hh  = (const float*)d_in[5];
    const float* g_ih  = (const float*)d_in[6];
    const float* b_ih  = (const float*)d_in[7];
    const float* g_hh  = (const float*)d_in[8];
    const float* b_hh  = (const float*)d_in[9];
    const float* g_c   = (const float*)d_in[10];
    const float* b_c   = (const float*)d_in[11];
    const float* W_out = (const float*)d_in[12];
    const float* b_out = (const float*)d_in[13];

    float* out     = (float*)d_out;          // tuple elem 0: (128,256)
    float* emb_out = out + B_ * OUTN;        // tuple elem 1: (128,256,128)

    char* ws = (char*)d_ws;
    size_t off = 0;
    auto alloc = [&](size_t bytes) {
        void* p = ws + off;
        off = (off + bytes + 255) & ~(size_t)255;
        return p;
    };
    _Float16* emb16  = (_Float16*)alloc((size_t)B_ * T_ * D_ * sizeof(_Float16));
    _Float16* Wih16  = (_Float16*)alloc((size_t)G4_ * D_ * sizeof(_Float16));
    _Float16* Wout16 = (_Float16*)alloc((size_t)OUTN * KOUT * sizeof(_Float16));
    float*    xgbuf  = (float*)   alloc((size_t)B_ * T_ * G4_ * sizeof(float));
    _Float16* hs16   = (_Float16*)alloc((size_t)T_ * B_ * H_ * sizeof(_Float16));

    gather_embed_kernel<<<(B_ * T_ * 2 * 16) / 256, 256, 0, stream>>>(indices, table, emb_out, emb16);
    f32_to_f16_kernel<<<(G4_ * D_ + 255) / 256, 256, 0, stream>>>(W_ih, Wih16, G4_ * D_);
    f32_to_f16_kernel<<<(OUTN * KOUT + 255) / 256, 256, 0, stream>>>(W_out, Wout16, OUTN * KOUT);
    xgates_gemm_kernel<<<8192, 256, 0, stream>>>(emb16, Wih16, xgbuf);
    lstm_scan_kernel<<<8, 256, 0, stream>>>(xgbuf, W_hh, h0, c0,
                                            g_ih, b_ih, g_hh, b_hh, g_c, b_c, hs16);
    out_gemm_kernel<<<16, 256, 0, stream>>>(hs16, Wout16, b_out, out);
}
